// HNSNet_50500225466443
// MI455X (gfx1250) — matrix operand order repaired and checked by reading, not compiled, not run on hardware
//
#include <hip/hip_runtime.h>
#include <hip/hip_bf16.h>
#include <stdint.h>

// ---------------------------------------------------------------------------
// HNSNet forward for MI455X (gfx1250, wave32, WMMA).
//   uv_encoder  : conv1 (VALU fp32) -> conv2 as 9 shifted GEMMs via
//                 v_wmma_f32_16x16x32_f16 (single live accumulator, no spills)
//                 -> relu+avgpool fused -> fc 64->128  => x[:,0:128]
//   sym_encoder : 3->64->64 MLP  => x[:,128:192]
//   per GAT layer:
//     gemm_wmma : h = x @ W + b    (WMMA f16 tiles, W transposed in LDS)
//     node_scores, edge pass1 (logits + atomicMax), pass2 (denom),
//     pass3 (vectorized b128 gather + scatter-add), elu
//   classifier  : 128->64 relu ->25
// ---------------------------------------------------------------------------

typedef __attribute__((ext_vector_type(16))) _Float16 v16h;
typedef __attribute__((ext_vector_type(8)))  _Float16 v8h;
typedef __attribute__((ext_vector_type(8)))  float    v8f;

union H16 { v16h v; v8h h[2]; _Float16 e[16]; };

__device__ __forceinline__ unsigned encf(float f) {
    int i = __float_as_int(f);
    return (i < 0) ? ~((unsigned)i) : (((unsigned)i) | 0x80000000u);
}
__device__ __forceinline__ float decf(unsigned k) {
    int i = (k & 0x80000000u) ? (int)(k & 0x7fffffffu) : ~((int)k);
    return __int_as_float(i);
}

__global__ void fillf_kernel(float* p, unsigned n, float v) {
    unsigned i = blockIdx.x * 256u + threadIdx.x;
    if (i < n) p[i] = v;
}
__global__ void fillu_kernel(unsigned* p, unsigned n, unsigned v) {
    unsigned i = blockIdx.x * 256u + threadIdx.x;
    if (i < n) p[i] = v;
}

// ---------------------------------------------------------------------------
// Symbolic encoder: x_sym[N,3] -> relu(fc 3->64) -> relu(fc 64->64)
__global__ void sym_encode_kernel(const float* __restrict__ xs,
                                  const float* __restrict__ w1, const float* __restrict__ b1,
                                  const float* __restrict__ w2, const float* __restrict__ b2,
                                  float* __restrict__ xout, int n) {
    int node = blockIdx.x * blockDim.x + threadIdx.x;
    if (node >= n) return;
    float x0 = xs[node * 3 + 0], x1 = xs[node * 3 + 1], x2 = xs[node * 3 + 2];
    float h1[64];
#pragma unroll
    for (int j = 0; j < 64; ++j)
        h1[j] = fmaxf(b1[j] + x0 * w1[j] + x1 * w1[64 + j] + x2 * w1[128 + j], 0.f);
    for (int j = 0; j < 64; ++j) {
        float a = b2[j];
#pragma unroll
        for (int k = 0; k < 64; ++k) a += h1[k] * w2[k * 64 + j];
        xout[(size_t)node * 192 + 128 + j] = fmaxf(a, 0.f);
    }
}

// ---------------------------------------------------------------------------
// UV encoder. Block = 128 threads = 4 waves; one wave per node (grid-stride).
__global__ void uv_encode_kernel(const float* __restrict__ xvis,
                                 const float* __restrict__ c1w, const float* __restrict__ c1b,
                                 const float* __restrict__ c2w, const float* __restrict__ c2b,
                                 const float* __restrict__ fcw, const float* __restrict__ fcb,
                                 float* __restrict__ xout, int n) {
    __shared__ float    sC1W[1728];                      // [oc=32][ic=6][3][3]
    __shared__ float    sC1B[32];
    __shared__ __align__(32) _Float16 sC2W[9 * 64 * 32]; // [shift][oc][ic] f16
    __shared__ float    sC2B[64];
    __shared__ __align__(16) _Float16 sH1[4][64 * 32];   // per wave: [pixel][ic] f16
    __shared__ float    sPool[4][64];

    const int tid = threadIdx.x;
    for (int i = tid; i < 1728; i += 128) sC1W[i] = c1w[i];
    for (int i = tid; i < 32; i += 128) sC1B[i] = c1b[i];
    for (int i = tid; i < 64; i += 128) sC2B[i] = c2b[i];
    for (int i = tid; i < 9 * 64 * 32; i += 128) {
        int s = i / 2048, r = i - s * 2048;
        int oc = r >> 5, ic = r & 31;
        sC2W[i] = (_Float16)c2w[oc * 288 + ic * 9 + s];   // [O][I][3][3], s=ky*3+kx
    }
    __syncthreads();

    const int wv = tid >> 5, lane = tid & 31;
    const int lm = lane & 15, lh = lane >> 4;
    _Float16* h1 = sH1[wv];

    for (int node = blockIdx.x * 4 + wv; node < n; node += gridDim.x * 4) {
        const float* xv = xvis + (size_t)node * 384;   // [6][8][8]

        // ---- conv1 (6->32) + relu, fp32, 2 pixels per lane ----
        for (int pp = 0; pp < 2; ++pp) {
            int p = lane + pp * 32, py = p >> 3, px = p & 7;
            float patch[54];
#pragma unroll
            for (int i = 0; i < 6; ++i)
#pragma unroll
                for (int ky = 0; ky < 3; ++ky)
#pragma unroll
                    for (int kx = 0; kx < 3; ++kx) {
                        int ys = py + ky - 1, xs = px + kx - 1;
                        patch[(i * 3 + ky) * 3 + kx] =
                            (ys >= 0 && ys < 8 && xs >= 0 && xs < 8) ? xv[i * 64 + ys * 8 + xs] : 0.f;
                    }
            for (int oc = 0; oc < 32; ++oc) {
                float a = sC1B[oc];
                const float* w = &sC1W[oc * 54];
#pragma unroll
                for (int t = 0; t < 54; ++t) a += patch[t] * w[t];
                h1[p * 32 + oc] = (_Float16)fmaxf(a, 0.f);
            }
        }
        asm volatile("s_wait_dscnt 0x0" ::: "memory");
        __builtin_amdgcn_wave_barrier();

        // ---- conv2 (32->64) as 9 shifted WMMA GEMMs, fused relu+avgpool ----
        // Per M-tile: preload 9 A fragments (read-only), then one accumulator
        // at a time over N-tiles -> low register pressure, no scratch spills.
        float psum[4] = {0.f, 0.f, 0.f, 0.f};
        for (int i = 0; i < 4; ++i) {
            int p = i * 16 + lm, py = p >> 3, px = p & 7;
            H16 Af[9];
#pragma unroll
            for (int s = 0; s < 9; ++s) {
                int ys = py + s / 3 - 1, xs = px + s % 3 - 1;
                if (ys >= 0 && ys < 8 && xs >= 0 && xs < 8) {
                    int q = ys * 8 + xs, kb = lh * 16;
                    Af[s].h[0] = *(const v8h*)&h1[q * 32 + kb];
                    Af[s].h[1] = *(const v8h*)&h1[q * 32 + kb + 8];
                } else {
#pragma unroll
                    for (int t = 0; t < 16; ++t) Af[s].e[t] = (_Float16)0.f;
                }
            }
            for (int j = 0; j < 4; ++j) {
                v8f acc;
#pragma unroll
                for (int r = 0; r < 8; ++r) acc[r] = 0.f;
                const int oc = j * 16 + lm, ks = lh * 16;
#pragma unroll
                for (int s = 0; s < 9; ++s) {
                    v16h B = *(const v16h*)&sC2W[s * 2048 + oc * 32 + ks];
                    acc = __builtin_amdgcn_wmma_f32_16x16x32_f16(
                        false, Af[s].v, false, B, (short)0, acc, false, false);
                }
                float bsb = sC2B[oc];
#pragma unroll
                for (int r = 0; r < 8; ++r) psum[j] += fmaxf(acc[r] + bsb, 0.f);
            }
        }
#pragma unroll
        for (int j = 0; j < 4; ++j) {
            float tot = psum[j] + __shfl_down(psum[j], 16);
            if (lane < 16) sPool[wv][j * 16 + lane] = tot * (1.f / 64.f);
        }
        asm volatile("s_wait_dscnt 0x0" ::: "memory");
        __builtin_amdgcn_wave_barrier();

        // ---- fc 64->128 into x[:,0:128] ----
        for (int t = 0; t < 4; ++t) {
            int o = lane + t * 32;
            float a = fcb[o];
            for (int k = 0; k < 64; ++k) a += sPool[wv][k] * fcw[k * 128 + o];
            xout[(size_t)node * 192 + o] = a;
        }
        __builtin_amdgcn_wave_barrier();
    }
}

// ---------------------------------------------------------------------------
// WMMA GEMM: C[M,128] = A[M,K] @ W[K,128] + bias.  K in {128,192}.
__global__ void gemm_wmma_kernel(const float* __restrict__ A, const float* __restrict__ W,
                                 const float* __restrict__ bias, float* __restrict__ C,
                                 int M, int K) {
    __shared__ __align__(32) _Float16 sW[128 * 192];   // transposed: [n][k]
    const int tid = threadIdx.x;
    for (int idx = tid; idx < 128 * K; idx += 256) {
        int nn = idx / K, k = idx - nn * K;
        sW[nn * K + k] = (_Float16)W[(size_t)k * 128 + nn];
    }
    __syncthreads();

    const int wv = tid >> 5, lane = tid & 31;
    const int lm = lane & 15, lh = lane >> 4;
    const int rowBase = blockIdx.x * 128 + wv * 16;
    const int m = rowBase + lm;
    const bool inm = (m < M);
    const float* ap = A + (size_t)(inm ? m : 0) * K;

    v8f acc[8];
#pragma unroll
    for (int j = 0; j < 8; ++j)
#pragma unroll
        for (int r = 0; r < 8; ++r) acc[j][r] = 0.f;

    for (int k0 = 0; k0 < K; k0 += 32) {
        H16 Af;
        int kb = k0 + lh * 16;
        __builtin_prefetch(ap + kb + 32, 0, 1);
#pragma unroll
        for (int t = 0; t < 8; ++t) {
            Af.e[t]     = (_Float16)(inm ? ap[kb + t] : 0.f);
            Af.e[8 + t] = (_Float16)(inm ? ap[kb + 8 + t] : 0.f);
        }
#pragma unroll
        for (int j = 0; j < 8; ++j) {
            int nn = j * 16 + lm, ks = k0 + lh * 16;
            v16h B = *(const v16h*)&sW[nn * K + ks];
            acc[j] = __builtin_amdgcn_wmma_f32_16x16x32_f16(
                false, Af.v, false, B, (short)0, acc[j], false, false);
        }
    }
#pragma unroll
    for (int j = 0; j < 8; ++j) {
        int nn = j * 16 + lm;
        float bs = bias[nn];
#pragma unroll
        for (int r = 0; r < 8; ++r) {
            int mr = rowBase + lh * 8 + r;
            if (mr < M) C[(size_t)mr * 128 + nn] = acc[j][r] + bs;
        }
    }
}

// ---------------------------------------------------------------------------
__global__ void node_scores_kernel(const float* __restrict__ h,
                                   const float* __restrict__ asrc, const float* __restrict__ adst,
                                   float* __restrict__ ssrc, float* __restrict__ sdst, int n) {
    int idx = blockIdx.x * blockDim.x + threadIdx.x;
    if (idx >= n * 4) return;
    int node = idx >> 2, hh = idx & 3;
    const float* hp = &h[(size_t)node * 128 + hh * 32];
    float s1 = 0.f, s2 = 0.f;
#pragma unroll
    for (int d = 0; d < 32; ++d) { s1 += hp[d] * asrc[hh * 32 + d]; s2 += hp[d] * adst[hh * 32 + d]; }
    ssrc[idx] = s1; sdst[idx] = s2;
}

// prep: pr[L*12 + j*4+h] = We_j . a_edge_h ; pr[L*12+8+h] = be . a_edge_h
__global__ void prep_kernel(const float* We1, const float* be1, const float* ae1,
                            const float* We2, const float* be2, const float* ae2,
                            float* pr) {
    int t = threadIdx.x;
    if (t >= 24) return;
    int L = t / 12, r = t % 12;
    const float* We = L ? We2 : We1;
    const float* be = L ? be2 : be1;
    const float* ae = L ? ae2 : ae1;
    float acc = 0.f;
    if (r < 8) {
        int j = r / 4, hh = r % 4;
        for (int d = 0; d < 32; ++d) acc += We[j * 128 + hh * 32 + d] * ae[hh * 32 + d];
    } else {
        int hh = r - 8;
        for (int d = 0; d < 32; ++d) acc += be[hh * 32 + d] * ae[hh * 32 + d];
    }
    pr[t] = acc;
}

// Edge pass 1: logits + segment-max (order-preserving uint atomicMax).
__global__ void edge_logits_kernel(const int* __restrict__ ei, const float* __restrict__ ea,
                                   const float* __restrict__ ssrc, const float* __restrict__ sdst,
                                   const float* __restrict__ pr, float* __restrict__ logits,
                                   unsigned* __restrict__ mEnc, int E) {
    int e = blockIdx.x * 256 + threadIdx.x;
    if (e >= E) return;
    int s = ei[e], d = ei[E + e];
    float a0 = ea[2 * e], a1 = ea[2 * e + 1];
#pragma unroll
    for (int h = 0; h < 4; ++h) {
        float l = ssrc[s * 4 + h] + sdst[d * 4 + h] + a0 * pr[h] + a1 * pr[4 + h] + pr[8 + h];
        l = (l > 0.f) ? l : 0.2f * l;
        logits[(size_t)e * 4 + h] = l;
        atomicMax(&mEnc[d * 4 + h], encf(l));
    }
}

// Edge pass 2: softmax denominators.
__global__ void edge_denom_kernel(const int* __restrict__ ei, const float* __restrict__ logits,
                                  const unsigned* __restrict__ mEnc, float* __restrict__ den, int E) {
    int e = blockIdx.x * 256 + threadIdx.x;
    if (e >= E) return;
    int d = ei[E + e];
#pragma unroll
    for (int h = 0; h < 4; ++h) {
        float ex = __expf(logits[(size_t)e * 4 + h] - decf(mEnc[d * 4 + h]));
        atomicAdd(&den[d * 4 + h], ex);
    }
}

// Edge pass 3: msg = (h_src + e_feat) * alpha * gate, scatter-add to out[dst].
// One thread per (edge, 4 channels): a wave covers exactly one edge, so all
// per-edge scalars are wave-uniform; h_src/We/be move as b128 loads.
__global__ void edge_scatter_kernel(const int* __restrict__ ei, const float* __restrict__ ea,
                                    const float* __restrict__ h, const float* __restrict__ logits,
                                    const unsigned* __restrict__ mEnc, const float* __restrict__ den,
                                    const float* __restrict__ We, const float* __restrict__ be,
                                    const float* __restrict__ wg, const float* __restrict__ bg,
                                    float* __restrict__ out, int E) {
    unsigned idx = blockIdx.x * 256u + threadIdx.x;
    if (idx >= (unsigned)E * 32u) return;
    int e = idx >> 5;
    int c0 = (idx & 31) * 4;           // 4 consecutive channels
    int hh = c0 >> 5;
    int s = ei[e], d = ei[E + e];
    float a0 = ea[2 * e], a1 = ea[2 * e + 1];
    float l = logits[(size_t)e * 4 + hh];
    float alpha = __expf(l - decf(mEnc[d * 4 + hh])) / (den[d * 4 + hh] + 1e-16f);
    float gate = 1.f / (1.f + __expf(-(a0 * wg[0] + a1 * wg[1] + bg[0])));
    float ag = alpha * gate;

    const float4 hs = *(const float4*)&h[(size_t)s * 128 + c0];
    const float4 w0 = *(const float4*)&We[c0];
    const float4 w1 = *(const float4*)&We[128 + c0];
    const float4 bb = *(const float4*)&be[c0];
    float* op = &out[(size_t)d * 128 + c0];
    atomicAdd(op + 0, (hs.x + a0 * w0.x + a1 * w1.x + bb.x) * ag);
    atomicAdd(op + 1, (hs.y + a0 * w0.y + a1 * w1.y + bb.y) * ag);
    atomicAdd(op + 2, (hs.z + a0 * w0.z + a1 * w1.z + bb.z) * ag);
    atomicAdd(op + 3, (hs.w + a0 * w0.w + a1 * w1.w + bb.w) * ag);
}

__global__ void elu_kernel(float* p, unsigned n) {
    unsigned i = blockIdx.x * 256u + threadIdx.x;
    if (i >= n) return;
    float v = p[i];
    p[i] = (v > 0.f) ? v : (__expf(v) - 1.f);
}

// Classifier: relu(x@w1+b1)@w2+b2, per-node thread.
__global__ void classifier_kernel(const float* __restrict__ x,
                                  const float* __restrict__ w1, const float* __restrict__ b1,
                                  const float* __restrict__ w2, const float* __restrict__ b2,
                                  float* __restrict__ out, int n) {
    int node = blockIdx.x * blockDim.x + threadIdx.x;
    if (node >= n) return;
    float hid[64];
#pragma unroll
    for (int j = 0; j < 64; ++j) hid[j] = b1[j];
    for (int k = 0; k < 128; ++k) {
        float xv = x[(size_t)node * 128 + k];
#pragma unroll
        for (int j = 0; j < 64; ++j) hid[j] += xv * w1[k * 64 + j];
    }
#pragma unroll
    for (int j = 0; j < 64; ++j) hid[j] = fmaxf(hid[j], 0.f);
    for (int o = 0; o < 25; ++o) {
        float a = b2[o];
#pragma unroll
        for (int j = 0; j < 64; ++j) a += hid[j] * w2[j * 25 + o];
        out[(size_t)node * 25 + o] = a;
    }
}

// ---------------------------------------------------------------------------
extern "C" void kernel_launch(void* const* d_in, const int* in_sizes, int n_in,
                              void* d_out, int out_size, void* d_ws, size_t ws_size,
                              hipStream_t stream) {
    const int N = in_sizes[1] / 3;       // x_sym [N,3]
    const int E = in_sizes[3] / 2;       // edge_attr [E,2]

    const float* x_vis  = (const float*)d_in[0];
    const float* x_sym  = (const float*)d_in[1];
    const int*   ei     = (const int*)  d_in[2];
    const float* ea     = (const float*)d_in[3];
    const float* c1w    = (const float*)d_in[4];
    const float* c1b    = (const float*)d_in[5];
    const float* c2w    = (const float*)d_in[6];
    const float* c2b    = (const float*)d_in[7];
    const float* uvfcw  = (const float*)d_in[8];
    const float* uvfcb  = (const float*)d_in[9];
    const float* s1w    = (const float*)d_in[10];
    const float* s1b    = (const float*)d_in[11];
    const float* s2w    = (const float*)d_in[12];
    const float* s2b    = (const float*)d_in[13];
    const float* g1W    = (const float*)d_in[14];
    const float* g1b    = (const float*)d_in[15];
    const float* g1We   = (const float*)d_in[16];
    const float* g1be   = (const float*)d_in[17];
    const float* g1as   = (const float*)d_in[18];
    const float* g1ad   = (const float*)d_in[19];
    const float* g1ae   = (const float*)d_in[20];
    const float* g1Wg   = (const float*)d_in[21];
    const float* g1bg   = (const float*)d_in[22];
    const float* g2W    = (const float*)d_in[23];
    const float* g2b    = (const float*)d_in[24];
    const float* g2We   = (const float*)d_in[25];
    const float* g2be   = (const float*)d_in[26];
    const float* g2as   = (const float*)d_in[27];
    const float* g2ad   = (const float*)d_in[28];
    const float* g2ae   = (const float*)d_in[29];
    const float* g2Wg   = (const float*)d_in[30];
    const float* g2bg   = (const float*)d_in[31];
    const float* c1wfc  = (const float*)d_in[32];
    const float* c1bfc  = (const float*)d_in[33];
    const float* c2wfc  = (const float*)d_in[34];
    const float* c2bfc  = (const float*)d_in[35];
    float* outp = (float*)d_out;

    // workspace layout (floats)
    float* ws = (float*)d_ws;
    float*    X0   = ws;                         // [N,192] features; later layer-2 output [N,128]
    float*    H    = X0 + (size_t)N * 192;       // [N,128] GAT transform
    float*    G    = H  + (size_t)N * 128;       // [N,128] GAT layer-1 output / layer-2 input
    float*    SS   = G  + (size_t)N * 128;       // [N,4]
    float*    SD   = SS + (size_t)N * 4;         // [N,4]
    unsigned* MEnc = (unsigned*)(SD + (size_t)N * 4); // [N,4]
    float*    DEN  = (float*)MEnc + (size_t)N * 4;    // [N,4]
    float*    LG   = DEN + (size_t)N * 4;        // [E,4]
    float*    PR   = LG + (size_t)E * 4;         // [24]

    auto blocks = [](long long n, int b) { return (unsigned)((n + b - 1) / b); };

    // encoders + attention-const prep
    prep_kernel<<<1, 32, 0, stream>>>(g1We, g1be, g1ae, g2We, g2be, g2ae, PR);
    sym_encode_kernel<<<blocks(N, 128), 128, 0, stream>>>(x_sym, s1w, s1b, s2w, s2b, X0, N);
    uv_encode_kernel<<<512, 128, 0, stream>>>(x_vis, c1w, c1b, c2w, c2b, uvfcw, uvfcb, X0, N);

    // ---- GAT layer 1 ----
    fillf_kernel<<<blocks((long long)N * 128, 256), 256, 0, stream>>>(G, (unsigned)(N * 128), 0.f);
    fillu_kernel<<<blocks((long long)N * 4, 256), 256, 0, stream>>>(MEnc, (unsigned)(N * 4), 0u);
    fillf_kernel<<<blocks((long long)N * 4, 256), 256, 0, stream>>>(DEN, (unsigned)(N * 4), 0.f);
    gemm_wmma_kernel<<<blocks(N, 128), 256, 0, stream>>>(X0, g1W, g1b, H, N, 192);
    node_scores_kernel<<<blocks((long long)N * 4, 256), 256, 0, stream>>>(H, g1as, g1ad, SS, SD, N);
    edge_logits_kernel<<<blocks(E, 256), 256, 0, stream>>>(ei, ea, SS, SD, PR, LG, MEnc, E);
    edge_denom_kernel<<<blocks(E, 256), 256, 0, stream>>>(ei, LG, MEnc, DEN, E);
    edge_scatter_kernel<<<blocks((long long)E * 32, 256), 256, 0, stream>>>(
        ei, ea, H, LG, MEnc, DEN, g1We, g1be, g1Wg, g1bg, G, E);
    elu_kernel<<<blocks((long long)N * 128, 256), 256, 0, stream>>>(G, (unsigned)(N * 128));

    // ---- GAT layer 2 (output into X0 reused as [N,128]) ----
    fillf_kernel<<<blocks((long long)N * 128, 256), 256, 0, stream>>>(X0, (unsigned)(N * 128), 0.f);
    fillu_kernel<<<blocks((long long)N * 4, 256), 256, 0, stream>>>(MEnc, (unsigned)(N * 4), 0u);
    fillf_kernel<<<blocks((long long)N * 4, 256), 256, 0, stream>>>(DEN, (unsigned)(N * 4), 0.f);
    gemm_wmma_kernel<<<blocks(N, 128), 256, 0, stream>>>(G, g2W, g2b, H, N, 128);
    node_scores_kernel<<<blocks((long long)N * 4, 256), 256, 0, stream>>>(H, g2as, g2ad, SS, SD, N);
    edge_logits_kernel<<<blocks(E, 256), 256, 0, stream>>>(ei, ea, SS, SD, PR + 12, LG, MEnc, E);
    edge_denom_kernel<<<blocks(E, 256), 256, 0, stream>>>(ei, LG, MEnc, DEN, E);
    edge_scatter_kernel<<<blocks((long long)E * 32, 256), 256, 0, stream>>>(
        ei, ea, H, LG, MEnc, DEN, g2We, g2be, g2Wg, g2bg, X0, E);
    elu_kernel<<<blocks((long long)N * 128, 256), 256, 0, stream>>>(X0, (unsigned)(N * 128));

    // ---- classifier ----
    classifier_kernel<<<blocks(N, 128), 128, 0, stream>>>(X0, c1wfc, c1bfc, c2wfc, c2bfc, outp, N);
}
